// MashDecoder_59098749993428
// MI455X (gfx1250) — compile-verified, matrix operand order checked by reading, not run on hardware
//
#include <hip/hip_runtime.h>
#include <hip/hip_bf16.h>
#include <math.h>

// ---------------------------------------------------------------------------
// MashDecoder forward for MI455X (gfx1250, wave32, WMMA).
// Dense GEMMs -> v_wmma_f32_16x16x32_bf16 (f32 accum), LDS-staged, 8 WMMA/iter.
// ---------------------------------------------------------------------------

#define B_      8
#define L_      400
#define NQ_     8192
#define DMODEL  400
#define DINNER  800
#define DSTATE  16
#define DCONV   4
#define DTRANK  25
#define NLAYER  24
#define HEADS   4
#define DHEAD   400
#define INNER_  1600
#define FFHID   1600
#define EPS_    1e-5f

typedef __bf16 bf16_t;
typedef __attribute__((ext_vector_type(16))) __bf16 v16bf;
typedef __attribute__((ext_vector_type(8)))  __bf16 v8bf;
typedef __attribute__((ext_vector_type(8)))  float  v8f;

__device__ __forceinline__ unsigned bfbits(float f) {
  union { float f; unsigned u; } v; v.f = f;
  return (v.u + 0x7FFFu + ((v.u >> 16) & 1u)) >> 16;
}
__device__ __forceinline__ bf16_t f2bf(float f) {
  unsigned short hs = (unsigned short)bfbits(f);
  return __builtin_bit_cast(bf16_t, hs);
}
__device__ __forceinline__ unsigned pkbf(float a, float b) {
  return (bfbits(a) & 0xFFFFu) | (bfbits(b) << 16);
}

// ---------------------------------------------------------------------------
// Generic WMMA GEMM: C[M,N] = op( alpha * (A[M,K] @ B) + bias )   (+= if ACC)
//   A row-major (lda). B row-major KxN (ldb), or transB: B[k][n] = Bp[n*ldb+k].
// flags: 1=bias[n], 2=softplus, 4=alpha-scale, 8=accumulate into C.
// Block 256 thr = 8 waves (2 x 4); wave tile 64x32 = 4x2 WMMA tiles.
// Block tile 128x128, BK=32.  Interior blocks: vectorized unguarded staging.
// Boundary: branchless clamped staging (cndmask select, no exec branches).
// ---------------------------------------------------------------------------
#define GF_BIAS 1
#define GF_SOFTPLUS 2
#define GF_SCALE 4
#define GF_ACC 8

__global__ __launch_bounds__(256) void gemm_wmma_bf16(
    const float* __restrict__ A, int lda,
    const float* __restrict__ Bp, int ldb, int transB,
    float* __restrict__ C, int ldc,
    int M, int N, int K,
    const float* __restrict__ bias, int flags, float alpha)
{
  __shared__ __bf16 sA[128 * 32];  // [m][k], k contiguous
  __shared__ __bf16 sB[128 * 32];  // transposed: [n][k], k contiguous

  const int tid   = threadIdx.x;
  const int lane  = tid & 31;
  const int wave  = tid >> 5;       // 0..7
  const int wm    = wave >> 2;      // 0..1 -> 64-row half
  const int wn    = wave & 3;       // 0..3 -> 32-col strip
  const int tileM = blockIdx.y * 128;
  const int tileN = blockIdx.x * 128;

  const bool interior = (tileM + 128 <= M) && (tileN + 128 <= N);
  const bool aVec  = ((lda & 3) == 0);
  const bool bVec  = ((ldb & 3) == 0);

  v8f acc[4][2] = {};

  const int mhalf = lane & 15;
  const int kb0   = (lane < 16) ? 0 : 8;   // per-lane K base (16-bit A layout)

  for (int k0 = 0; k0 < K; k0 += 32) {
    const bool fullK = (k0 + 32 <= K);
    if (interior && fullK) {
      // ---------------- fast path: unguarded, vectorized ----------------
      if (aVec) {
#pragma unroll
        for (int i = 0; i < 4; ++i) {
          int chunk = tid + i * 256;            // 1024 chunks of 4
          int r = chunk >> 3, c4 = (chunk & 7) * 4;
          float4 f = *(const float4*)(A + (size_t)(tileM + r) * lda + k0 + c4);
          uint2 p; p.x = pkbf(f.x, f.y); p.y = pkbf(f.z, f.w);
          *(uint2*)(sA + r * 32 + c4) = p;
        }
      } else {
#pragma unroll
        for (int i = 0; i < 16; ++i) {
          int idx = tid + i * 256;              // 4096 elems
          int r = idx >> 5, c = idx & 31;
          sA[idx] = f2bf(A[(size_t)(tileM + r) * lda + k0 + c]);
        }
      }
      if (!transB) {
        // gather 4 k's per (n) column, pack, one b64 LDS store
#pragma unroll
        for (int i = 0; i < 4; ++i) {
          int chunk = tid + i * 256;
          int n = chunk & 127, k4 = (chunk >> 7) * 4;
          const float* bp = Bp + (size_t)(k0 + k4) * ldb + tileN + n;
          uint2 p;
          p.x = pkbf(bp[0], bp[(size_t)ldb]);
          p.y = pkbf(bp[2 * (size_t)ldb], bp[3 * (size_t)ldb]);
          *(uint2*)(sB + n * 32 + k4) = p;
        }
      } else if (bVec) {
#pragma unroll
        for (int i = 0; i < 4; ++i) {
          int chunk = tid + i * 256;
          int n = chunk >> 3, k4 = (chunk & 7) * 4;
          float4 f = *(const float4*)(Bp + (size_t)(tileN + n) * ldb + k0 + k4);
          uint2 p; p.x = pkbf(f.x, f.y); p.y = pkbf(f.z, f.w);
          *(uint2*)(sB + n * 32 + k4) = p;
        }
      } else {
#pragma unroll
        for (int i = 0; i < 16; ++i) {
          int idx = tid + i * 256;
          int n = idx >> 5, k = idx & 31;
          sB[n * 32 + k] = f2bf(Bp[(size_t)(tileN + n) * ldb + k0 + k]);
        }
      }
      if (k0 + 32 < K)
        __builtin_prefetch(A + (size_t)(tileM + (tid >> 1)) * lda + k0 + 32, 0, 1);
    } else {
      // ---------------- boundary path: branchless clamped ----------------
#pragma unroll
      for (int i = 0; i < 16; ++i) {
        int idx = tid + i * 256;
        int r = idx >> 5, c = idx & 31;
        int gr = tileM + r, gc = k0 + c;
        int grc = min(gr, M - 1), gcc = min(gc, K - 1);
        float v = A[(size_t)grc * lda + gcc];
        v = (gr < M && gc < K) ? v : 0.f;
        sA[idx] = f2bf(v);
      }
#pragma unroll
      for (int i = 0; i < 16; ++i) {
        int idx = tid + i * 256;
        int k = idx >> 7, n = idx & 127;
        int gk = k0 + k, gn = tileN + n;
        int gkc = min(gk, K - 1), gnc = min(gn, N - 1);
        float v = transB ? Bp[(size_t)gnc * ldb + gkc] : Bp[(size_t)gkc * ldb + gnc];
        v = (gk < K && gn < N) ? v : 0.f;
        sB[n * 32 + k] = f2bf(v);
      }
    }
    __syncthreads();

    // ---- fragments: two 16B LDS reads each, per documented 16-bit layout ----
    v16bf aF[4], bF[2];
#pragma unroll
    for (int i = 0; i < 4; ++i) {
      int m = wm * 64 + i * 16 + mhalf;
      v8bf lo = *(const v8bf*)(sA + m * 32 + kb0);
      v8bf hi = *(const v8bf*)(sA + m * 32 + kb0 + 16);
      aF[i] = __builtin_shufflevector(lo, hi, 0,1,2,3,4,5,6,7,8,9,10,11,12,13,14,15);
    }
#pragma unroll
    for (int j = 0; j < 2; ++j) {
      int n = wn * 32 + j * 16 + mhalf;
      v8bf lo = *(const v8bf*)(sB + n * 32 + kb0);
      v8bf hi = *(const v8bf*)(sB + n * 32 + kb0 + 16);
      bF[j] = __builtin_shufflevector(lo, hi, 0,1,2,3,4,5,6,7,8,9,10,11,12,13,14,15);
    }
#pragma unroll
    for (int i = 0; i < 4; ++i)
#pragma unroll
      for (int j = 0; j < 2; ++j)
        acc[i][j] = __builtin_amdgcn_wmma_f32_16x16x32_bf16(
            false, aF[i], false, bF[j], (short)0, acc[i][j], false, false);
    __syncthreads();
  }

  // ---- epilogue: C/D layout: vgpr r -> M = mbase + r, N = lane & 15 ----
  const int mbase = (lane < 16) ? 0 : 8;
  const int ncol  = lane & 15;
  if (interior) {
#pragma unroll
    for (int i = 0; i < 4; ++i) {
#pragma unroll
      for (int j = 0; j < 2; ++j) {
        int gn = tileN + wn * 32 + j * 16 + ncol;
        int gm0 = tileM + wm * 64 + i * 16 + mbase;
        float bv = (flags & GF_BIAS) ? bias[gn] : 0.f;
#pragma unroll
        for (int r = 0; r < 8; ++r) {
          float v = acc[i][j][r];
          if (flags & GF_SCALE) v *= alpha;
          v += bv;
          if (flags & GF_SOFTPLUS) v = (v > 20.f) ? v : log1pf(expf(v));
          size_t o = (size_t)(gm0 + r) * ldc + gn;
          if (flags & GF_ACC) v += C[o];
          C[o] = v;
        }
      }
    }
  } else {
#pragma unroll
    for (int i = 0; i < 4; ++i) {
#pragma unroll
      for (int j = 0; j < 2; ++j) {
        int gn = tileN + wn * 32 + j * 16 + ncol;
        if (gn >= N) continue;
        int gm0 = tileM + wm * 64 + i * 16 + mbase;
        float bv = (flags & GF_BIAS) ? bias[gn] : 0.f;
#pragma unroll
        for (int r = 0; r < 8; ++r) {
          int gm = gm0 + r;
          if (gm >= M) continue;
          float v = acc[i][j][r];
          if (flags & GF_SCALE) v *= alpha;
          v += bv;
          if (flags & GF_SOFTPLUS) v = (v > 20.f) ? v : log1pf(expf(v));
          size_t o = (size_t)gm * ldc + gn;
          if (flags & GF_ACC) v += C[o];
          C[o] = v;
        }
      }
    }
  }
}

// ---------------------------------------------------------------------------
// Elementwise / reduction kernels
// ---------------------------------------------------------------------------

__global__ __launch_bounds__(256) void transpose_mash(const float* __restrict__ mp,
                                                      float* __restrict__ x) {
  int idx = blockIdx.x * 256 + threadIdx.x;
  if (idx >= B_ * L_ * DMODEL) return;
  int dm = idx % DMODEL; int t = idx / DMODEL;
  int l = t % L_; int b = t / L_;
  x[idx] = mp[((size_t)b * DMODEL + dm) * L_ + l];
}

__global__ __launch_bounds__(256) void rmsnorm_rows(const float* __restrict__ x,
                                                    const float* __restrict__ w,
                                                    float* __restrict__ out, int cols) {
  int row = blockIdx.x;
  __shared__ float red[256];
  float s = 0.f;
  for (int c = threadIdx.x; c < cols; c += 256) { float v = x[(size_t)row * cols + c]; s += v * v; }
  red[threadIdx.x] = s; __syncthreads();
  for (int off = 128; off > 0; off >>= 1) {
    if (threadIdx.x < off) red[threadIdx.x] += red[threadIdx.x + off];
    __syncthreads();
  }
  float inv = rsqrtf(red[0] / (float)cols + EPS_);
  for (int c = threadIdx.x; c < cols; c += 256)
    out[(size_t)row * cols + c] = x[(size_t)row * cols + c] * inv * w[c];
}

__global__ __launch_bounds__(256) void ln_rows(const float* __restrict__ x,
                                               const float* __restrict__ w,
                                               const float* __restrict__ b,
                                               float* __restrict__ out, int cols) {
  int row = blockIdx.x;
  __shared__ float r1[256], r2[256];
  float s = 0.f, s2 = 0.f;
  for (int c = threadIdx.x; c < cols; c += 256) {
    float v = x[(size_t)row * cols + c]; s += v; s2 += v * v;
  }
  r1[threadIdx.x] = s; r2[threadIdx.x] = s2; __syncthreads();
  for (int off = 128; off > 0; off >>= 1) {
    if (threadIdx.x < off) { r1[threadIdx.x] += r1[threadIdx.x + off]; r2[threadIdx.x] += r2[threadIdx.x + off]; }
    __syncthreads();
  }
  float m = r1[0] / (float)cols;
  float var = r2[0] / (float)cols - m * m;
  float inv = rsqrtf(var + EPS_);
  __syncthreads();
  for (int c = threadIdx.x; c < cols; c += 256)
    out[(size_t)row * cols + c] = (x[(size_t)row * cols + c] - m) * inv * w[c] + b[c];
}

__global__ __launch_bounds__(256) void softmax_rows(float* __restrict__ x, int cols) {
  int row = blockIdx.x;
  __shared__ float red[256];
  float mx = -1e30f;
  for (int c = threadIdx.x; c < cols; c += 256) mx = fmaxf(mx, x[(size_t)row * cols + c]);
  red[threadIdx.x] = mx; __syncthreads();
  for (int off = 128; off > 0; off >>= 1) {
    if (threadIdx.x < off) red[threadIdx.x] = fmaxf(red[threadIdx.x], red[threadIdx.x + off]);
    __syncthreads();
  }
  float mval = red[0]; __syncthreads();
  float s = 0.f;
  for (int c = threadIdx.x; c < cols; c += 256) {
    float e = expf(x[(size_t)row * cols + c] - mval);
    x[(size_t)row * cols + c] = e; s += e;
  }
  red[threadIdx.x] = s; __syncthreads();
  for (int off = 128; off > 0; off >>= 1) {
    if (threadIdx.x < off) red[threadIdx.x] += red[threadIdx.x + off];
    __syncthreads();
  }
  float inv = 1.f / red[0];
  for (int c = threadIdx.x; c < cols; c += 256) x[(size_t)row * cols + c] *= inv;
}

// depthwise causal conv (k=4) over L on first DINNER cols of xz, then silu
__global__ __launch_bounds__(256) void conv_silu(const float* __restrict__ xz,
                                                 const float* __restrict__ cw,
                                                 const float* __restrict__ cb,
                                                 float* __restrict__ xca) {
  int idx = blockIdx.x * 256 + threadIdx.x;
  if (idx >= B_ * L_ * DINNER) return;
  int d = idx % DINNER; int t = idx / DINNER;
  int l = t % L_; int b = t / L_;
  float s = cb[d];
#pragma unroll
  for (int k = 0; k < DCONV; ++k) {
    int ls = l + k - (DCONV - 1);
    if (ls >= 0) s += xz[((size_t)(b * L_ + ls)) * (2 * DINNER) + d] * cw[d * DCONV + k];
  }
  xca[idx] = s / (1.f + expf(-s));   // silu
}

// selective scan: one thread per (b,d); fuses dA*h+dB*x, C-dot, +D*x, silu(z) gate
__global__ __launch_bounds__(256) void mamba_scan(const float* __restrict__ dt,
                                                  const float* __restrict__ xca,
                                                  const float* __restrict__ xdbl,
                                                  const float* __restrict__ xz,
                                                  const float* __restrict__ A_log,
                                                  const float* __restrict__ Dw,
                                                  float* __restrict__ y) {
  int idx = blockIdx.x * 256 + threadIdx.x;
  if (idx >= B_ * DINNER) return;
  int b = idx / DINNER, d = idx % DINNER;
  float As[DSTATE];
#pragma unroll
  for (int s = 0; s < DSTATE; ++s) As[s] = -expf(A_log[d * DSTATE + s]);
  float Dv = Dw[d];
  float h[DSTATE];
#pragma unroll
  for (int s = 0; s < DSTATE; ++s) h[s] = 0.f;
  for (int l = 0; l < L_; ++l) {
    int r = b * L_ + l;
    float dtv = dt[(size_t)r * DINNER + d];
    float xcv = xca[(size_t)r * DINNER + d];
    float zv  = xz[(size_t)r * (2 * DINNER) + DINNER + d];
    const float* xd = xdbl + (size_t)r * (DTRANK + 2 * DSTATE);
    float accv = 0.f;
#pragma unroll
    for (int s = 0; s < DSTATE; ++s) {
      float Bs = xd[DTRANK + s];
      float Cs = xd[DTRANK + DSTATE + s];
      h[s] = expf(dtv * As[s]) * h[s] + dtv * Bs * xcv;
      accv += h[s] * Cs;
    }
    float sz = zv / (1.f + expf(-zv));
    y[(size_t)r * DINNER + d] = (accv + Dv * xcv) * sz;
  }
}

// build [sin(q@basis), cos(q@basis), q] features (51 per row)
__global__ __launch_bounds__(256) void build_feat(const float* __restrict__ qry,
                                                  const float* __restrict__ basis,
                                                  float* __restrict__ feat) {
  int row = blockIdx.x * 256 + threadIdx.x;
  if (row >= B_ * NQ_) return;
  float q0 = qry[row * 3], q1 = qry[row * 3 + 1], q2 = qry[row * 3 + 2];
  float* f = feat + (size_t)row * 51;
#pragma unroll
  for (int j = 0; j < 24; ++j) {
    float p = q0 * basis[j] + q1 * basis[24 + j] + q2 * basis[48 + j];
    f[j] = sinf(p);
    f[24 + j] = cosf(p);
  }
  f[48] = q0; f[49] = q1; f[50] = q2;
}

// ag = a * gelu_exact(g);  a = hb[:, :1600], g = hb[:, 1600:]
__global__ __launch_bounds__(256) void gelu_mul(const float* __restrict__ hb,
                                                float* __restrict__ ag, int rows) {
  int idx = blockIdx.x * 256 + threadIdx.x;
  if (idx >= rows * FFHID) return;
  int m = idx / FFHID, j = idx % FFHID;
  float a = hb[(size_t)m * (2 * FFHID) + j];
  float g = hb[(size_t)m * (2 * FFHID) + FFHID + j];
  float ge = 0.5f * g * (1.f + erff(g * 0.70710678118654752f));
  ag[idx] = a * ge;
}

// out[row] = lat[row,:] . Wout + bout   (wave32 per row)
// Wout staged to LDS via CDNA5 async global->LDS copy (ASYNCcnt path).
__global__ __launch_bounds__(256) void out_dot(const float* __restrict__ lat,
                                               const float* __restrict__ Wout,
                                               const float* __restrict__ bout,
                                               float* __restrict__ out, int rows) {
  __shared__ __align__(16) float sw[DMODEL];
  int tid = threadIdx.x;
  int c4 = tid * 4;
  if (c4 < DMODEL) {   // 100 lanes x 16B = 1600B
    unsigned lds_addr = (unsigned)(uintptr_t)(&sw[c4]);
    unsigned long long gaddr = (unsigned long long)(uintptr_t)(Wout + c4);
    asm volatile("global_load_async_to_lds_b128 %0, %1, off"
                 :: "v"(lds_addr), "v"(gaddr) : "memory");
  }
  asm volatile("s_wait_asynccnt 0" ::: "memory");
  __syncthreads();

  int wid = (blockIdx.x * 256 + tid) >> 5;
  int lane = tid & 31;
  if (wid >= rows) return;
  float s = 0.f;
  for (int c = lane; c < DMODEL; c += 32) s += lat[(size_t)wid * DMODEL + c] * sw[c];
#pragma unroll
  for (int off = 16; off > 0; off >>= 1) s += __shfl_down(s, off, 32);
  if (lane == 0) out[wid] = s + bout[0];
}

// ---------------------------------------------------------------------------
// Host orchestration
// ---------------------------------------------------------------------------

static inline float* carve(void* ws, size_t& off, size_t count) {
  float* p = (float*)((char*)ws + off);
  off += count * sizeof(float);
  off = (off + 255) & ~(size_t)255;
  return p;
}

static void gemm(hipStream_t st, const float* A, int lda, const float* Bm, int ldb,
                 int transB, float* C, int ldc, int M, int N, int K,
                 const float* bias, int flags, float alpha) {
  dim3 grid((N + 127) / 128, (M + 127) / 128);
  gemm_wmma_bf16<<<grid, 256, 0, st>>>(A, lda, Bm, ldb, transB, C, ldc, M, N, K,
                                       bias, flags, alpha);
}

extern "C" void kernel_launch(void* const* d_in, const int* in_sizes, int n_in,
                              void* d_out, int out_size, void* d_ws, size_t ws_size,
                              hipStream_t stream) {
  // setup_inputs() dict insertion order, recursive:
  const float* mash    = (const float*)d_in[0];   // (8,400,400)
  const float* qry     = (const float*)d_in[1];   // (8,8192,3)
  const float* norm_w  = (const float*)d_in[2];   // (24,400)
  const float* in_proj = (const float*)d_in[3];   // (24,400,1600)
  const float* conv_w  = (const float*)d_in[4];   // (24,800,4)
  const float* conv_b  = (const float*)d_in[5];   // (24,800)
  const float* x_proj  = (const float*)d_in[6];   // (24,800,57)
  const float* dt_w    = (const float*)d_in[7];   // (24,25,800)
  const float* dt_b    = (const float*)d_in[8];   // (24,800)
  const float* A_log   = (const float*)d_in[9];   // (24,800,16)
  const float* Dp      = (const float*)d_in[10];  // (24,800)
  const float* out_pj  = (const float*)d_in[11];  // (24,800,400)
  const float* basis   = (const float*)d_in[12];  // (3,24)
  const float* pe_w    = (const float*)d_in[13];  // (51,400)
  const float* pe_b    = (const float*)d_in[14];  // (400)
  const float* q_ln_w  = (const float*)d_in[15];
  const float* q_ln_b  = (const float*)d_in[16];
  const float* c_ln_w  = (const float*)d_in[17];
  const float* c_ln_b  = (const float*)d_in[18];
  const float* Wq      = (const float*)d_in[19];  // (400,1600)
  const float* Wkv     = (const float*)d_in[20];  // (400,3200)
  const float* Wo      = (const float*)d_in[21];  // (1600,400)
  const float* bo      = (const float*)d_in[22];  // (400)
  const float* f_ln_w  = (const float*)d_in[23];
  const float* f_ln_b  = (const float*)d_in[24];
  const float* W1      = (const float*)d_in[25];  // (400,3200)
  const float* b1      = (const float*)d_in[26];  // (3200)
  const float* W2      = (const float*)d_in[27];  // (1600,400)
  const float* b2      = (const float*)d_in[28];  // (400)
  const float* Wout    = (const float*)d_in[29];  // (400,1)
  const float* bout    = (const float*)d_in[30];  // (1)
  float* out = (float*)d_out;                     // (8,8192)
  (void)in_sizes; (void)n_in; (void)out_size; (void)ws_size;

  const int ROWS = B_ * L_;        // 3200
  const int QROWS = B_ * NQ_;      // 65536

  size_t off = 0;
  float* x    = carve(d_ws, off, (size_t)ROWS * DMODEL);
  float* xn   = carve(d_ws, off, (size_t)ROWS * DMODEL);
  float* xz   = carve(d_ws, off, (size_t)ROWS * 2 * DINNER);
  float* xca  = carve(d_ws, off, (size_t)ROWS * DINNER);
  float* xdbl = carve(d_ws, off, (size_t)ROWS * (DTRANK + 2 * DSTATE));
  float* dtb  = carve(d_ws, off, (size_t)ROWS * DINNER);
  float* ybuf = carve(d_ws, off, (size_t)ROWS * DINNER);
  float* cn   = carve(d_ws, off, (size_t)ROWS * DMODEL);
  float* kv   = carve(d_ws, off, (size_t)ROWS * 2 * INNER_);
  float* feat = carve(d_ws, off, (size_t)QROWS * 51);
  float* qe   = carve(d_ws, off, (size_t)QROWS * DMODEL);
  float* lat  = carve(d_ws, off, (size_t)QROWS * DMODEL);
  float* qb   = carve(d_ws, off, (size_t)NQ_ * INNER_);
  float* att  = carve(d_ws, off, (size_t)NQ_ * L_);
  float* ob   = carve(d_ws, off, (size_t)NQ_ * INNER_);
  float* hln  = carve(d_ws, off, (size_t)NQ_ * DMODEL);
  float* hb   = carve(d_ws, off, (size_t)NQ_ * 2 * FFHID);
  float* ag   = carve(d_ws, off, (size_t)NQ_ * FFHID);

  // x = mash.transpose(0,2,1)
  transpose_mash<<<(ROWS * DMODEL + 255) / 256, 256, 0, stream>>>(mash, x);

  // ---- Mamba stack ----
  for (int l = 0; l < NLAYER; ++l) {
    rmsnorm_rows<<<ROWS, 256, 0, stream>>>(x, norm_w + (size_t)l * DMODEL, xn, DMODEL);
    gemm(stream, xn, DMODEL, in_proj + (size_t)l * DMODEL * 2 * DINNER, 2 * DINNER, 0,
         xz, 2 * DINNER, ROWS, 2 * DINNER, DMODEL, nullptr, 0, 1.f);
    conv_silu<<<(ROWS * DINNER + 255) / 256, 256, 0, stream>>>(
        xz, conv_w + (size_t)l * DINNER * DCONV, conv_b + (size_t)l * DINNER, xca);
    gemm(stream, xca, DINNER, x_proj + (size_t)l * DINNER * (DTRANK + 2 * DSTATE),
         DTRANK + 2 * DSTATE, 0, xdbl, DTRANK + 2 * DSTATE,
         ROWS, DTRANK + 2 * DSTATE, DINNER, nullptr, 0, 1.f);
    gemm(stream, xdbl, DTRANK + 2 * DSTATE, dt_w + (size_t)l * DTRANK * DINNER, DINNER, 0,
         dtb, DINNER, ROWS, DINNER, DTRANK,
         dt_b + (size_t)l * DINNER, GF_BIAS | GF_SOFTPLUS, 1.f);
    mamba_scan<<<(B_ * DINNER + 255) / 256, 256, 0, stream>>>(
        dtb, xca, xdbl, xz, A_log + (size_t)l * DINNER * DSTATE, Dp + (size_t)l * DINNER, ybuf);
    gemm(stream, ybuf, DINNER, out_pj + (size_t)l * DINNER * DMODEL, DMODEL, 0,
         x, DMODEL, ROWS, DMODEL, DINNER, nullptr, GF_ACC, 1.f);
  }

  // ---- Cross attention ----
  ln_rows<<<ROWS, 256, 0, stream>>>(x, c_ln_w, c_ln_b, cn, DMODEL);
  gemm(stream, cn, DMODEL, Wkv, 2 * INNER_, 0, kv, 2 * INNER_,
       ROWS, 2 * INNER_, DMODEL, nullptr, 0, 1.f);

  build_feat<<<(QROWS + 255) / 256, 256, 0, stream>>>(qry, basis, feat);
  gemm(stream, feat, 51, pe_w, DMODEL, 0, qe, DMODEL, QROWS, DMODEL, 51,
       pe_b, GF_BIAS, 1.f);
  ln_rows<<<QROWS, 256, 0, stream>>>(qe, q_ln_w, q_ln_b, qe, DMODEL);  // qn in place

  const float scale = 0.05f;  // 400^-0.5
  for (int b = 0; b < B_; ++b) {
    const float* qn_b = qe + (size_t)b * NQ_ * DMODEL;
    gemm(stream, qn_b, DMODEL, Wq, INNER_, 0, qb, INNER_, NQ_, INNER_, DMODEL,
         nullptr, 0, 1.f);
    for (int h = 0; h < HEADS; ++h) {
      gemm(stream, qb + h * DHEAD, INNER_,
           kv + (size_t)b * L_ * 2 * INNER_ + h * DHEAD, 2 * INNER_, 1,
           att, L_, NQ_, L_, DHEAD, nullptr, GF_SCALE, scale);
      softmax_rows<<<NQ_, 256, 0, stream>>>(att, L_);
      gemm(stream, att, L_,
           kv + (size_t)b * L_ * 2 * INNER_ + INNER_ + h * DHEAD, 2 * INNER_, 0,
           ob + h * DHEAD, INNER_, NQ_, DHEAD, L_, nullptr, 0, 1.f);
    }
    float* lat_b = lat + (size_t)b * NQ_ * DMODEL;
    gemm(stream, ob, INNER_, Wo, DMODEL, 0, lat_b, DMODEL, NQ_, DMODEL, INNER_,
         bo, GF_BIAS, 1.f);
    ln_rows<<<NQ_, 256, 0, stream>>>(lat_b, f_ln_w, f_ln_b, hln, DMODEL);
    gemm(stream, hln, DMODEL, W1, 2 * FFHID, 0, hb, 2 * FFHID, NQ_, 2 * FFHID, DMODEL,
         b1, GF_BIAS, 1.f);
    gelu_mul<<<(NQ_ * FFHID + 255) / 256, 256, 0, stream>>>(hb, ag, NQ_);
    gemm(stream, ag, FFHID, W2, DMODEL, 0, lat_b, DMODEL, NQ_, DMODEL, FFHID,
         b2, GF_BIAS | GF_ACC, 1.f);
  }

  // out = lat @ Wout + bout
  out_dot<<<(QROWS * 32 + 255) / 256, 256, 0, stream>>>(lat, Wout, bout, out, QROWS);
}
